// GCN_1623497638612
// MI455X (gfx1250) — matrix operand order repaired and checked by reading, not compiled, not run on hardware
//
#include <hip/hip_runtime.h>
#include <hip/hip_bf16.h>

typedef _Float16 v16h __attribute__((ext_vector_type(16)));
typedef float    v8f  __attribute__((ext_vector_type(8)));
typedef unsigned int v8u __attribute__((ext_vector_type(8)));

#define HDIM 128
#define CDIM 10

__device__ __forceinline__ unsigned int pack_h2(float a, float b) {
  unsigned short lo = __builtin_bit_cast(unsigned short, (_Float16)a);
  unsigned short hi = __builtin_bit_cast(unsigned short, (_Float16)b);
  return ((unsigned int)hi << 16) | (unsigned int)lo;
}

// ---- zero a float region ----
__global__ void k_zero(float* __restrict__ p, long long n) {
  long long i = (long long)blockIdx.x * blockDim.x + threadIdx.x;
  if (i < n) p[i] = 0.0f;
}

// ---- pack W [128 in][128 out] fp32 -> WMMA B-fragment half2 layout ----
// wp[ntile*1024 + kb*256 + j*32 + lane] = half2(W[k][n], W[k+1][n])
// with m=lane&15, hi=lane>>4, k = kb*32 + hi*16 + 2*j, n = ntile*16 + m
__global__ void k_pack_w(const float* __restrict__ W, unsigned int* __restrict__ wp) {
  int tid = blockIdx.x * blockDim.x + threadIdx.x;   // 0..8191
  int lane  = tid & 31;
  int j     = (tid >> 5) & 7;
  int kb    = (tid >> 8) & 3;
  int ntile = tid >> 10;
  int m = lane & 15, hi = lane >> 4;
  int k = kb * 32 + hi * 16 + 2 * j;
  int n = ntile * 16 + m;
  wp[tid] = pack_h2(W[k * HDIM + n], W[(k + 1) * HDIM + n]);
}

// ---- in-degree count (PyG GCNConv: deg from col) ----
__global__ void k_deg(const int* __restrict__ col, float* __restrict__ deg, int E) {
  int e = blockIdx.x * blockDim.x + threadIdx.x;
  if (e < E) atomicAdd(&deg[col[e]], 1.0f);
}

// ---- dinv = rsqrt(deg + 1) (self-loop adds 1; always > 0) ----
__global__ void k_dinv(float* __restrict__ deg, int N) {
  int n = blockIdx.x * blockDim.x + threadIdx.x;
  if (n < N) deg[n] = rsqrtf(deg[n] + 1.0f);
}

// ---- Out[N,128] = X[N,128] @ Wpack (WMMA f32<-f16 16x16x32) ----
__global__ __launch_bounds__(256) void k_gemm(const float* __restrict__ X,
                                              const unsigned int* __restrict__ Wpack,
                                              float* __restrict__ Out, int nrows) {
  __shared__ unsigned int sA[16 * 66];         // 16 rows x 128 halfs, pitch 66 dwords
  const int mbase = blockIdx.x * 16;
  const int t = threadIdx.x;
  {
    int r = t >> 4;
    int c0 = (t & 15) * 8;
    int grow = mbase + r;
    unsigned int* dst = &sA[r * 66 + (c0 >> 1)];
    if (grow < nrows) {
      const float4* src = (const float4*)(X + (long long)grow * HDIM + c0);
      float4 f0 = src[0];
      float4 f1 = src[1];
      dst[0] = pack_h2(f0.x, f0.y);
      dst[1] = pack_h2(f0.z, f0.w);
      dst[2] = pack_h2(f1.x, f1.y);
      dst[3] = pack_h2(f1.z, f1.w);
    } else {
      dst[0] = 0u; dst[1] = 0u; dst[2] = 0u; dst[3] = 0u;
    }
  }
  __syncthreads();

  const int lane = t & 31;
  const int wave = t >> 5;                     // ntile 0..7
  const int m = lane & 15, hi = lane >> 4;
  v8f acc = {0.f, 0.f, 0.f, 0.f, 0.f, 0.f, 0.f, 0.f};
  const unsigned int* wp = Wpack + wave * 1024 + lane;

#pragma unroll
  for (int kb = 0; kb < 4; ++kb) {
    v8u au, bu;
#pragma unroll
    for (int j = 0; j < 8; ++j) {
      int k = kb * 32 + hi * 16 + 2 * j;
      au[j] = sA[m * 66 + (k >> 1)];
      bu[j] = wp[kb * 256 + j * 32];
    }
    v16h a = __builtin_bit_cast(v16h, au);
    v16h b = __builtin_bit_cast(v16h, bu);
    acc = __builtin_amdgcn_wmma_f32_16x16x32_f16(false, a, false, b,
                                                 (short)0, acc, false, false);
  }

  const int colx = wave * 16 + m;
  if (mbase + 16 <= nrows) {
    // uniform fast path: 8 plain stores with immediate offsets
    float* o = Out + (long long)(mbase + hi * 8) * HDIM + colx;
#pragma unroll
    for (int j = 0; j < 8; ++j) o[j * HDIM] = acc[j];
  } else {
#pragma unroll
    for (int j = 0; j < 8; ++j) {
      int rowg = mbase + hi * 8 + j;
      if (rowg < nrows) Out[(long long)rowg * HDIM + colx] = acc[j];
    }
  }
}

// ---- gather-scale-scatter over E edges + N self loops; 32 lanes/edge x float4 ----
__global__ void k_scatter(const float* __restrict__ src, float* __restrict__ dst,
                          const int* __restrict__ row, const int* __restrict__ col,
                          const float* __restrict__ dinv, int E, int N) {
  long long g = (long long)blockIdx.x * blockDim.x + threadIdx.x;
  long long e = g >> 5;                 // wave-uniform: 32 lanes per edge
  int d = (int)(g & 31) * 4;
  if (e >= (long long)E + N) return;    // uniform per wave
  int r, c;
  if (e < E) { r = row[e]; c = col[e]; }
  else       { r = c = (int)(e - E); }
  // indices are wave-uniform -> push index math / norm loads to scalar path
  r = __builtin_amdgcn_readfirstlane(r);
  c = __builtin_amdgcn_readfirstlane(c);
  float nrm = dinv[r] * dinv[c];
  const float4 v = *(const float4*)(src + (long long)r * HDIM + d);
  float* o = dst + (long long)c * HDIM + d;
  atomicAdd(o + 0, v.x * nrm);
  atomicAdd(o + 1, v.y * nrm);
  atomicAdd(o + 2, v.z * nrm);
  atomicAdd(o + 3, v.w * nrm);
}

// ---- h = relu(h + b) ----
__global__ void k_bias_relu(float* __restrict__ h, const float* __restrict__ b,
                            long long total) {
  long long i = (long long)blockIdx.x * blockDim.x + threadIdx.x;
  if (i < total) {
    float v = h[i] + b[(int)(i & (HDIM - 1))];
    h[i] = v > 0.0f ? v : 0.0f;
  }
}

// ---- mean-pool accumulation: 32 lanes/node x float4 ----
__global__ void k_pool(const float* __restrict__ h, const int* __restrict__ batch,
                       float* __restrict__ pooled, float* __restrict__ cnt, int N) {
  long long g = (long long)blockIdx.x * blockDim.x + threadIdx.x;
  long long n = g >> 5;                 // wave-uniform
  int d = (int)(g & 31) * 4;
  if (n >= N) return;
  int b = __builtin_amdgcn_readfirstlane(batch[n]);
  const float4 v = *(const float4*)(h + n * HDIM + d);
  float* o = pooled + (long long)b * HDIM + d;
  atomicAdd(o + 0, v.x);
  atomicAdd(o + 1, v.y);
  atomicAdd(o + 2, v.z);
  atomicAdd(o + 3, v.w);
  if (d == 0) atomicAdd(&cnt[b], 1.0f);
}

// ---- head: mean, [G,128]@[128,10]+b, log_softmax ----
__global__ void k_head(const float* __restrict__ pooled, const float* __restrict__ cnt,
                       const float* __restrict__ Wout, const float* __restrict__ bout,
                       float* __restrict__ out, int G) {
  int g = blockIdx.x * blockDim.x + threadIdx.x;
  if (g >= G) return;
  float inv = 1.0f / fmaxf(cnt[g], 1.0f);
  float acc[CDIM];
#pragma unroll
  for (int c = 0; c < CDIM; ++c) acc[c] = bout[c];
  for (int d = 0; d < HDIM; ++d) {
    float m = pooled[(long long)g * HDIM + d] * inv;
#pragma unroll
    for (int c = 0; c < CDIM; ++c) acc[c] += m * Wout[d * CDIM + c];
  }
  float mx = acc[0];
#pragma unroll
  for (int c = 1; c < CDIM; ++c) mx = fmaxf(mx, acc[c]);
  float s = 0.0f;
#pragma unroll
  for (int c = 0; c < CDIM; ++c) s += __expf(acc[c] - mx);
  float lse = mx + __logf(s);
#pragma unroll
  for (int c = 0; c < CDIM; ++c) out[g * CDIM + c] = acc[c] - lse;
}

extern "C" void kernel_launch(void* const* d_in, const int* in_sizes, int n_in,
                              void* d_out, int out_size, void* d_ws, size_t ws_size,
                              hipStream_t stream) {
  const float* x    = (const float*)d_in[0];
  const float* W1   = (const float*)d_in[1];
  const float* b1   = (const float*)d_in[2];
  const float* W2   = (const float*)d_in[3];
  const float* b2   = (const float*)d_in[4];
  const float* Wout = (const float*)d_in[5];
  const float* bout = (const float*)d_in[6];
  const int*   eidx = (const int*)d_in[7];
  const int*   batch= (const int*)d_in[8];

  const int N = in_sizes[8];
  const int E = in_sizes[7] / 2;
  const int G = out_size / CDIM;
  const int* row = eidx;
  const int* col = eidx + E;

  float* ws = (float*)d_ws;
  long long off = 0;
  float* dinv   = ws + off; off += N;
  float* t1     = ws + off; off += (long long)N * HDIM;
  float* pooled = ws + off; off += (long long)G * HDIM;
  float* cnt    = ws + off; off += G;
  long long zeroCount = off;              // deg+t1+pooled+cnt contiguous
  float* t0     = ws + off; off += (long long)N * HDIM;
  unsigned int* wp1 = (unsigned int*)(ws + off); off += 8192;
  unsigned int* wp2 = (unsigned int*)(ws + off); off += 8192;
  (void)ws_size; (void)n_in;

  const int TB = 256;
  const long long tot = (long long)N * HDIM;
  const int mtiles = (N + 15) / 16;

  // zero accumulation buffers (deg, t1, pooled, cnt)
  k_zero<<<(unsigned)((zeroCount + TB - 1) / TB), TB, 0, stream>>>(ws, zeroCount);
  // pack weights to f16 WMMA-B fragments
  k_pack_w<<<32, TB, 0, stream>>>(W1, wp1);
  k_pack_w<<<32, TB, 0, stream>>>(W2, wp2);
  // degrees -> dinv
  k_deg<<<(E + TB - 1) / TB, TB, 0, stream>>>(col, dinv, E);
  k_dinv<<<(N + TB - 1) / TB, TB, 0, stream>>>(dinv, N);

  // layer 1
  k_gemm<<<mtiles, TB, 0, stream>>>(x, wp1, t0, N);
  long long sthreads = ((long long)(E + N)) * 32;
  k_scatter<<<(unsigned)((sthreads + TB - 1) / TB), TB, 0, stream>>>(t0, t1, row, col, dinv, E, N);
  k_bias_relu<<<(unsigned)((tot + TB - 1) / TB), TB, 0, stream>>>(t1, b1, tot);

  // layer 2
  k_gemm<<<mtiles, TB, 0, stream>>>(t1, wp2, t0, N);
  k_zero<<<(unsigned)((tot + TB - 1) / TB), TB, 0, stream>>>(t1, tot);
  k_scatter<<<(unsigned)((sthreads + TB - 1) / TB), TB, 0, stream>>>(t0, t1, row, col, dinv, E, N);
  k_bias_relu<<<(unsigned)((tot + TB - 1) / TB), TB, 0, stream>>>(t1, b2, tot);

  // mean pool + head
  long long pthreads = (long long)N * 32;
  k_pool<<<(unsigned)((pthreads + TB - 1) / TB), TB, 0, stream>>>(t1, batch, pooled, cnt, N);
  k_head<<<(G + TB - 1) / TB, TB, 0, stream>>>(pooled, cnt, Wout, bout, (float*)d_out, G);
}
